// MixtralAttention_87686052315717
// MI455X (gfx1250) — matrix operand order repaired and checked by reading, not compiled, not run on hardware
//
#include <hip/hip_runtime.h>
#include <hip/hip_bf16.h>

typedef __attribute__((ext_vector_type(16))) __bf16 v16bf;
typedef __attribute__((ext_vector_type(8)))  __bf16 v8bf;
typedef __attribute__((ext_vector_type(4)))  __bf16 v4bf;
typedef __attribute__((ext_vector_type(8)))  float  v8f;

#define SHUF16(lo, hi) __builtin_shufflevector((lo), (hi), 0,1,2,3,4,5,6,7,8,9,10,11,12,13,14,15)

__device__ __forceinline__ v8f wmma_bf16(v16bf a, v16bf b, v8f c) {
  // (neg_a, A, neg_b, B, c_mod, C, reuse_a, reuse_b)
  return __builtin_amdgcn_wmma_f32_16x16x32_bf16(false, a, false, b, (short)0, c, false, false);
}

// ---------------------------------------------------------------------------
// f32 -> bf16 bulk convert (vectorized, bandwidth bound)
// ---------------------------------------------------------------------------
__global__ __launch_bounds__(256) void cvt_f32_bf16(const float4* __restrict__ in,
                                                    v4bf* __restrict__ out, size_t n4) {
  size_t i = (size_t)blockIdx.x * blockDim.x + threadIdx.x;
  const size_t stride = (size_t)gridDim.x * blockDim.x;
  for (; i < n4; i += stride) {
    float4 f = in[i];
    v4bf o = { (__bf16)f.x, (__bf16)f.y, (__bf16)f.z, (__bf16)f.w };
    out[i] = o;
  }
}

// ---------------------------------------------------------------------------
// C[M,N] = A[M,K] @ W[N,K]^T ; A,W bf16 row-major, C f32.
// One wave computes a 32x128 tile (2 M-subtiles x 8 N-subtiles,
// 16 WMMA per K-step against 20 b128 loads).
// ---------------------------------------------------------------------------
__global__ __launch_bounds__(256) void gemm_bf16(const __bf16* __restrict__ A,
                                                 const __bf16* __restrict__ W,
                                                 float* __restrict__ C,
                                                 int M, int N, int K) {
  const int lane = threadIdx.x & 31;
  const int lm = lane & 15;
  const int lh = lane >> 4;
  const int gw = (int)((blockIdx.x * blockDim.x + threadIdx.x) >> 5);
  const int wavesN = N >> 7;
  if (gw >= (M >> 5) * wavesN) return;
  const int m0 = (gw / wavesN) << 5;
  const int n0 = (gw % wavesN) << 7;

  const __bf16* a0p = A + (size_t)(m0 + lm) * K;
  const __bf16* a1p = A + (size_t)(m0 + 16 + lm) * K;
  const __bf16* bp[8];
#pragma unroll
  for (int j = 0; j < 8; ++j) bp[j] = W + (size_t)(n0 + 16 * j + lm) * K + 16 * lh;

  v8f acc[2][8] = {};
  for (int k = 0; k < K; k += 32) {
    __builtin_prefetch(a0p + k + 1024, 0, 3);
    __builtin_prefetch(bp[0] + k + 1024, 0, 3);
    __builtin_prefetch(bp[4] + k + 1024, 0, 3);
    v16bf a0 = *(const v16bf*)(a0p + k + 16 * lh);
    v16bf a1 = *(const v16bf*)(a1p + k + 16 * lh);
#pragma unroll
    for (int j = 0; j < 8; ++j) {
      v16bf b = *(const v16bf*)(bp[j] + k);
      acc[0][j] = wmma_bf16(a0, b, acc[0][j]);
      acc[1][j] = wmma_bf16(a1, b, acc[1][j]);
    }
  }
#pragma unroll
  for (int i = 0; i < 2; ++i)
#pragma unroll
    for (int j = 0; j < 8; ++j) {
      float* cp = C + (size_t)(m0 + 16 * i + 8 * lh) * N + n0 + 16 * j + lm;
#pragma unroll
      for (int r = 0; r < 8; ++r) cp[(size_t)r * N] = acc[i][j][r];
    }
}

// ---------------------------------------------------------------------------
// RoPE + scale + relayout:  qkv f32 [S,6144] ->
//   qT bf16 [32][S][128] (RoPE, * D^-1/2), kT bf16 [8][S][128] (RoPE),
//   vT bf16 [8][128][S]  (transposed for contiguous B-fragments in P@V)
// ---------------------------------------------------------------------------
__global__ __launch_bounds__(128) void rope_kernel(const int* __restrict__ positions,
                                                   const float* __restrict__ qkv,
                                                   __bf16* __restrict__ qT,
                                                   __bf16* __restrict__ kT,
                                                   __bf16* __restrict__ vT) {
  const int s = blockIdx.x;
  const int hd = blockIdx.y;   // 0..31 q heads, 32..39 k heads, 40..47 v heads
  const int d = threadIdx.x;   // 0..127
  const float* row = qkv + (size_t)s * 6144;
  if (hd < 40) {
    const int base = hd * 128;                 // q then k are contiguous
    const float x  = row[base + d];
    const float xo = row[base + (d ^ 64)];
    const int   i  = d & 63;
    const float pos = (float)positions[s];
    const float ang = pos * __expf(-0.21586735246819f * (float)i); // theta^(-i/64)
    float sn, cs;
    __sincosf(ang, &sn, &cs);
    const float out = (d < 64) ? (x * cs - xo * sn) : (x * cs + xo * sn);
    if (hd < 32)
      qT[((size_t)hd * 2048 + s) * 128 + d] = (__bf16)(out * 0.08838834764831845f);
    else
      kT[(((size_t)hd - 32) * 2048 + s) * 128 + d] = (__bf16)out;
  } else {
    const int h = hd - 40;
    vT[((size_t)h * 128 + d) * 2048 + s] = (__bf16)row[5120 + h * 128 + d];
  }
}

// ---------------------------------------------------------------------------
// Causal GQA flash attention. One wave owns a 16x128 Q tile of one head.
// Per 32-key block: 8 WMMA for Q@K^T, online softmax, 8 WMMA for P@V.
// P relayout C->A via wave-private LDS tile + s_wait_dscnt (same-wave in-order).
// ---------------------------------------------------------------------------
__global__ __launch_bounds__(128) void attn_kernel(const __bf16* __restrict__ qT,
                                                   const __bf16* __restrict__ kT,
                                                   const __bf16* __restrict__ vT,
                                                   __bf16* __restrict__ ao) {
  __shared__ __bf16 pbuf[4][16][32];
  const int lane = threadIdx.x & 31;
  const int wv = threadIdx.x >> 5;
  const int lm = lane & 15;
  const int lh = lane >> 4;
  const int gw = blockIdx.x * 4 + wv;
  const int head = gw >> 7;          // 128 q-tiles per head
  const int qt = gw & 127;
  const int kvh = head >> 2;         // 4 q heads per kv head
  const int q0 = qt << 4;

  const __bf16* qb = qT + ((size_t)head * 2048 + q0) * 128;
  const __bf16* kb = kT + (size_t)kvh * 2048 * 128;
  const __bf16* vb = vT + (size_t)kvh * 128 * 2048;
  __bf16 (*pt)[32] = pbuf[wv];

  v16bf qf[4];
#pragma unroll
  for (int d4 = 0; d4 < 4; ++d4) {
    const __bf16* p = qb + (size_t)lm * 128 + 32 * d4;
    qf[d4] = *(const v16bf*)(p + 16 * lh);
  }

  v8f acc[8] = {};
  float mrow[8], lrow[8];
#pragma unroll
  for (int r = 0; r < 8; ++r) { mrow[r] = -3e38f; lrow[r] = 0.f; }

  const int nkb = ((q0 + 15) >> 5) + 1;
  for (int kbk = 0; kbk < nkb; ++kbk) {
    const int kp0 = kbk << 5;
    v8f s0 = {}, s1 = {};
#pragma unroll
    for (int d4 = 0; d4 < 4; ++d4) {
      v16bf kf0 = *(const v16bf*)(kb + (size_t)(kp0 + lm) * 128 + 32 * d4 + 16 * lh);
      v16bf kf1 = *(const v16bf*)(kb + (size_t)(kp0 + 16 + lm) * 128 + 32 * d4 + 16 * lh);
      s0 = wmma_bf16(qf[d4], kf0, s0);
      s1 = wmma_bf16(qf[d4], kf1, s1);
    }
    float scl[8];
#pragma unroll
    for (int r = 0; r < 8; ++r) {
      const int qp = q0 + r + 8 * lh;
      float a = (kp0 + lm      <= qp) ? s0[r] : -1e30f;
      float b = (kp0 + 16 + lm <= qp) ? s1[r] : -1e30f;
      float mx = fmaxf(a, b);
      mx = fmaxf(mx, __shfl_xor(mx, 1));
      mx = fmaxf(mx, __shfl_xor(mx, 2));
      mx = fmaxf(mx, __shfl_xor(mx, 4));
      mx = fmaxf(mx, __shfl_xor(mx, 8));     // row lives in a 16-lane half
      const float mnew = fmaxf(mrow[r], mx);
      scl[r] = __expf(mrow[r] - mnew);
      const float p0 = __expf(a - mnew);
      const float p1 = __expf(b - mnew);
      float rs = p0 + p1;
      rs += __shfl_xor(rs, 1);
      rs += __shfl_xor(rs, 2);
      rs += __shfl_xor(rs, 4);
      rs += __shfl_xor(rs, 8);
      lrow[r] = lrow[r] * scl[r] + rs;
      mrow[r] = mnew;
      pt[r + 8 * lh][lm]      = (__bf16)p0;   // C-layout -> LDS
      pt[r + 8 * lh][16 + lm] = (__bf16)p1;
    }
#pragma unroll
    for (int t = 0; t < 8; ++t)
#pragma unroll
      for (int r = 0; r < 8; ++r) acc[t][r] *= scl[r];
    asm volatile("s_wait_dscnt 0" ::: "memory");   // same-wave LDS RAW
    v16bf pa = *(const v16bf*)&pt[lm][16 * lh];  // A-layout P
#pragma unroll
    for (int t = 0; t < 8; ++t) {
      v16bf vf = *(const v16bf*)(vb + (size_t)(16 * t + lm) * 2048 + kp0 + 16 * lh);
      acc[t] = wmma_bf16(pa, vf, acc[t]);
    }
    asm volatile("" ::: "memory");
  }
#pragma unroll
  for (int t = 0; t < 8; ++t)
#pragma unroll
    for (int r = 0; r < 8; ++r) {
      const int qp = q0 + r + 8 * lh;
      ao[(size_t)qp * 4096 + head * 128 + 16 * t + lm] = (__bf16)(acc[t][r] / lrow[r]);
    }
}

// ---------------------------------------------------------------------------
extern "C" void kernel_launch(void* const* d_in, const int* in_sizes, int n_in,
                              void* d_out, int out_size, void* d_ws, size_t ws_size,
                              hipStream_t stream) {
  const int*   positions = (const int*)d_in[0];
  const float* hs        = (const float*)d_in[1];
  const float* wqkv      = (const float*)d_in[2];
  const float* wo        = (const float*)d_in[3];

  char* w = (char*)d_ws;
  __bf16* hsb   = (__bf16*)w; w += (size_t)2048 * 4096 * 2;
  __bf16* wqkvb = (__bf16*)w; w += (size_t)6144 * 4096 * 2;
  __bf16* wob   = (__bf16*)w; w += (size_t)4096 * 4096 * 2;
  float*  qkv   = (float*)w;  w += (size_t)2048 * 6144 * 4;
  __bf16* qTb   = (__bf16*)w; w += (size_t)2048 * 4096 * 2;
  __bf16* kTb   = (__bf16*)w; w += (size_t)2048 * 1024 * 2;
  __bf16* vTb   = (__bf16*)w; w += (size_t)2048 * 1024 * 2;
  __bf16* aob   = (__bf16*)w; w += (size_t)2048 * 4096 * 2;

  cvt_f32_bf16<<<1024, 256, 0, stream>>>((const float4*)hs,   (v4bf*)hsb,   (size_t)2048 * 4096 / 4);
  cvt_f32_bf16<<<2048, 256, 0, stream>>>((const float4*)wqkv, (v4bf*)wqkvb, (size_t)6144 * 4096 / 4);
  cvt_f32_bf16<<<1024, 256, 0, stream>>>((const float4*)wo,   (v4bf*)wob,   (size_t)4096 * 4096 / 4);

  // qkv = hs @ w_qkv^T : M=2048 N=6144 K=4096 -> 64*48 = 3072 waves / 8 = 384 blocks
  gemm_bf16<<<384, 256, 0, stream>>>(hsb, wqkvb, qkv, 2048, 6144, 4096);

  rope_kernel<<<dim3(2048, 48), 128, 0, stream>>>(positions, qkv, qTb, kTb, vTb);

  // 32 heads * 128 q-tiles = 4096 waves / 4 = 1024 blocks
  attn_kernel<<<1024, 128, 0, stream>>>(qTb, kTb, vTb, aob);

  // out = attn_o @ w_o^T : M=2048 N=4096 K=4096 -> 64*32 = 2048 waves / 8 = 256 blocks
  gemm_bf16<<<256, 256, 0, stream>>>(aob, wob, (float*)d_out, 2048, 4096, 4096);
}
